// TGNMemoryWrapper_60421599920314
// MI455X (gfx1250) — compile-verified, matrix operand order checked by reading, not run on hardware
//
#include <hip/hip_runtime.h>
#include <math.h>

#define NN       100000
#define B_EV     16384
#define MEMD     512
#define MSGD     512
#define TIMED    128
#define EDGED    16
#define KMSG_RAW 1040          // 2*MEMD + EDGED
#define KMSG     1056          // padded to multiple of 32
#define XD       640           // MSGD + TIMED
#define G3       1536          // 3*MEMD

typedef __bf16 bf16;
typedef __attribute__((ext_vector_type(16))) __bf16 v16bf;
typedef __attribute__((ext_vector_type(8)))  __bf16 v8bf;
typedef __attribute__((ext_vector_type(8)))  float  v8f;

union FragU { v16bf v; v8bf h[2]; };

// Load one 16x32 bf16 WMMA operand fragment for this lane.
// Per ISA layout: lane&15 selects the row (M for A, N for B=W row), lane>>4
// selects K-half; elements 0..7 <- [kb + khalf*8, +8), 8..15 <- [kb+16+khalf*8, +8).
__device__ __forceinline__ v16bf frag_ld(const bf16* row, int kb, int khalf) {
  FragU f;
  f.h[0] = *(const v8bf*)(row + kb + khalf * 8);
  f.h[1] = *(const v8bf*)(row + kb + 16 + khalf * 8);
  return f.v;
}

__device__ __forceinline__ v8f wmma_bf16(v16bf a, v16bf b, v8f c) {
  return __builtin_amdgcn_wmma_f32_16x16x32_bf16(
      /*neg_a=*/false, a, /*neg_b=*/false, b,
      /*c_mod=*/(short)0, c, /*reuse_a=*/false, /*reuse_b=*/false);
}

__device__ __forceinline__ float sigmoidf_(float x) { return 1.0f / (1.0f + __expf(-x)); }

// Convert 8 f32 -> 8 bf16 and store as one 16B LDS write.
__device__ __forceinline__ void cvt_store8(bf16* d, float4 a, float4 b) {
  v8bf o;
  o[0] = (bf16)a.x; o[1] = (bf16)a.y; o[2] = (bf16)a.z; o[3] = (bf16)a.w;
  o[4] = (bf16)b.x; o[5] = (bf16)b.y; o[6] = (bf16)b.z; o[7] = (bf16)b.w;
  *(v8bf*)d = o;
}

// ---------------------------------------------------------------- weights->bf16
__global__ void k_convert_weights(const float* __restrict__ msgW,
                                  const float* __restrict__ Wih,
                                  const float* __restrict__ Whh,
                                  bf16* __restrict__ msgWb,
                                  bf16* __restrict__ Wihb,
                                  bf16* __restrict__ Whhb) {
  const int N1 = MSGD * KMSG;
  const int N2 = G3 * XD;
  const int N3 = G3 * MEMD;
  int i = blockIdx.x * blockDim.x + threadIdx.x;
  if (i >= N1 + N2 + N3) return;
  if (i < N1) {
    int r = i / KMSG, c = i % KMSG;
    msgWb[i] = (c < KMSG_RAW) ? (bf16)msgW[r * KMSG_RAW + c] : (bf16)0.0f;
  } else if (i < N1 + N2) {
    int j = i - N1; Wihb[j] = (bf16)Wih[j];
  } else {
    int j = i - N1 - N2; Whhb[j] = (bf16)Whh[j];
  }
}

// ---------------------------------------------------------------- scatter bookkeeping
__global__ void k_init_best(int* __restrict__ best) {
  int i = blockIdx.x * blockDim.x + threadIdx.x;
  if (i < NN) best[i] = -1;
}

__global__ void k_scatter_pos(const int* __restrict__ src, const int* __restrict__ tgt,
                              int* __restrict__ best) {
  int p = blockIdx.x * blockDim.x + threadIdx.x;
  if (p < 2 * B_EV) {
    int node = (p < B_EV) ? src[p] : tgt[p - B_EV];
    atomicMax(best + node, p);
  }
}

// ---------------------------------------------------------------- default outputs
__global__ void k_copy_mem(const float4* __restrict__ src, float4* __restrict__ dst, int n4) {
  int i = blockIdx.x * blockDim.x + threadIdx.x;
  int stride = gridDim.x * blockDim.x;
  for (; i < n4; i += stride) dst[i] = src[i];
}

__global__ void k_last_update(const float* __restrict__ ts, const float* __restrict__ lu,
                              const int* __restrict__ best, float* __restrict__ out_lu) {
  int i = blockIdx.x * blockDim.x + threadIdx.x;
  if (i < NN) {
    int b = best[i];
    out_lu[i] = (b >= 0) ? ts[b & (B_EV - 1)] : lu[i];
  }
}

// ---------------------------------------------------------------- time encoding -> x[:, 512:640]
__global__ void k_time_enc(const float* __restrict__ ts, const float* __restrict__ fr,
                           const float* __restrict__ ph, bf16* __restrict__ xbuf) {
  int i = blockIdx.x * blockDim.x + threadIdx.x;
  if (i < B_EV * TIMED) {
    int e = i >> 7, j = i & (TIMED - 1);
    xbuf[(size_t)e * XD + MSGD + j] = (bf16)cosf(ts[e] * fr[j] + ph[j]);
  }
}

// ---------------------------------------------------------------- message GEMM
// block: 64 events x 512 out cols; 8 waves; wave = 4 M-subtiles x 4 N-tiles.
// LDS rows interleaved ((r&15)*4 + r/16) so one wave's 4 A-subtile rows are
// adjacent -> all ds_load offsets fit the 16-bit immediate (no address VALU).
__global__ __launch_bounds__(256)
void k_msg_gemm(const int* __restrict__ src, const int* __restrict__ tgt,
                const float* __restrict__ edge, const float* __restrict__ mem,
                const bf16* __restrict__ Wb, const float* __restrict__ bias,
                bf16* __restrict__ xbuf) {
  extern __shared__ __align__(16) char smem[];
  bf16* As = (bf16*)smem;                       // [64][KMSG], row-permuted
  const int tile = blockIdx.x;
  const int lane = threadIdx.x, wave = threadIdx.y;
  const int tid = wave * 32 + lane;

  { // stage A rows: [src(512) | tgt(512) | edge(16) | zero-pad(16)] as bf16
    int r = tid >> 2, c4 = tid & 3;        // 64 rows, 4 threads/row
    int pr = (r & 15) * 4 + (r >> 4);      // interleaved LDS row
    int e = tile * 64 + r;
    const float4* ps4 = (const float4*)(mem + (size_t)src[e] * MEMD + c4 * 128);
    const float4* pt4 = (const float4*)(mem + (size_t)tgt[e] * MEMD + c4 * 128);
    bf16* row = As + pr * KMSG;
#pragma unroll
    for (int i = 0; i < 16; ++i)
      cvt_store8(row + c4 * 128 + i * 8, ps4[i * 2], ps4[i * 2 + 1]);
#pragma unroll
    for (int i = 0; i < 16; ++i)
      cvt_store8(row + 512 + c4 * 128 + i * 8, pt4[i * 2], pt4[i * 2 + 1]);
#pragma unroll
    for (int i = 0; i < 4; ++i) {
      row[1024 + c4 * 4 + i] = (bf16)edge[e * EDGED + c4 * 4 + i];
      row[1040 + c4 * 4 + i] = (bf16)0.0f;
    }
  }
  __syncthreads();

  const int m = lane & 15, khalf = lane >> 4;
  const bf16* abase = As + (m * 4) * KMSG;       // subtile es at +es*KMSG
  const v8f Z = {0,0,0,0,0,0,0,0};

#pragma unroll 1
  for (int j = 0; j < 4; ++j) {
    int n = (wave * 4 + j) * 16 + m;               // out col this lane owns
    const bf16* brow = Wb + (size_t)n * KMSG;
    v8f acc[4] = {Z, Z, Z, Z};
    for (int kb = 0; kb < KMSG; kb += 32) {
      v16bf bf = frag_ld(brow, kb, khalf);
      acc[0] = wmma_bf16(frag_ld(abase,            kb, khalf), bf, acc[0]);
      acc[1] = wmma_bf16(frag_ld(abase + KMSG,     kb, khalf), bf, acc[1]);
      acc[2] = wmma_bf16(frag_ld(abase + 2 * KMSG, kb, khalf), bf, acc[2]);
      acc[3] = wmma_bf16(frag_ld(abase + 3 * KMSG, kb, khalf), bf, acc[3]);
    }
    float bs = bias[n];
#pragma unroll
    for (int es = 0; es < 4; ++es) {
#pragma unroll
      for (int q = 0; q < 8; ++q) {
        int r0 = es * 16 + q + khalf * 8;
        float v = acc[es][q] + bs; v = v > 0.f ? v : 0.f;
        xbuf[(size_t)(tile * 64 + r0) * XD + n] = (bf16)v;
      }
    }
  }
}

// ---------------------------------------------------------------- fused GRU (gi + gh_src + gh_tgt + gates + winner scatter)
// LDS rows interleaved ((r&15)*2 + r/16) so both M-subtile rows sit at
// base and base+rowstride (immediate ds offsets).
__global__ __launch_bounds__(256)
void k_gru_fused(const int* __restrict__ src, const int* __restrict__ tgt,
                 const float* __restrict__ mem, const bf16* __restrict__ xbuf,
                 const bf16* __restrict__ Wihb, const bf16* __restrict__ Whhb,
                 const float* __restrict__ bih, const float* __restrict__ bhh,
                 const int* __restrict__ best, float* __restrict__ outMem) {
  extern __shared__ __align__(16) char smem[];
  bf16* Xs = (bf16*)smem;                 // [32][XD]   (filled by async copy)
  bf16* Hs = Xs + 32 * XD;                // [32][MEMD]
  bf16* Ht = Hs + 32 * MEMD;              // [32][MEMD]
  const int tile = blockIdx.x;
  const int lane = threadIdx.x, wave = threadIdx.y;
  const int tid = wave * 32 + lane;

  { // stage x via async global->LDS copy; gather+convert src/tgt rows via VALU
    int r = tid >> 3, c8 = tid & 7;
    int pr = (r & 15) * 2 + (r >> 4);     // interleaved LDS row
    int e = tile * 32 + r;
    // ---- async bf16 copy of x rows (ASYNCcnt-tracked, 16B per op per lane)
    unsigned lds_off = (unsigned)((pr * XD + c8 * 80) * sizeof(bf16));
    unsigned long long ga =
        (unsigned long long)(uintptr_t)(xbuf + (size_t)e * XD + c8 * 80);
#define ASYNC_CP16(OFF)                                                        \
    asm volatile("global_load_async_to_lds_b128 %0, %1, off offset:" #OFF     \
                 :: "v"(lds_off), "v"(ga) : "memory")
    ASYNC_CP16(0);   ASYNC_CP16(16);  ASYNC_CP16(32);  ASYNC_CP16(48);
    ASYNC_CP16(64);  ASYNC_CP16(80);  ASYNC_CP16(96);  ASYNC_CP16(112);
    ASYNC_CP16(128); ASYNC_CP16(144);
#undef ASYNC_CP16
    // ---- gathered memory rows f32 -> bf16
    const float4* ps4 = (const float4*)(mem + (size_t)src[e] * MEMD + c8 * 64);
    const float4* pt4 = (const float4*)(mem + (size_t)tgt[e] * MEMD + c8 * 64);
#pragma unroll
    for (int i = 0; i < 8; ++i)
      cvt_store8(Hs + pr * MEMD + c8 * 64 + i * 8, ps4[i * 2], ps4[i * 2 + 1]);
#pragma unroll
    for (int i = 0; i < 8; ++i)
      cvt_store8(Ht + pr * MEMD + c8 * 64 + i * 8, pt4[i * 2], pt4[i * 2 + 1]);
    asm volatile("s_wait_asynccnt 0x0" ::: "memory");
  }
  __syncthreads();

  const int m = lane & 15, khalf = lane >> 4;
  const bf16* xa = Xs + (m * 2) * XD;     // subtile es at +es*XD
  const bf16* sa = Hs + (m * 2) * MEMD;   // subtile es at +es*MEMD
  const bf16* ta = Ht + (m * 2) * MEMD;
  const v8f Z = {0,0,0,0,0,0,0,0};

#pragma unroll 1
  for (int j = 0; j < 4; ++j) {
    int n = (wave * 4 + j) * 16 + m;
    v8f gr[2] = {Z,Z}, gz[2] = {Z,Z}, gn[2] = {Z,Z};   // gi (shared src/tgt)
    v8f sr[2] = {Z,Z}, sz[2] = {Z,Z}, sv[2] = {Z,Z};   // gh src
    v8f tr[2] = {Z,Z}, tz[2] = {Z,Z}, tv[2] = {Z,Z};   // gh tgt

    { // gi = x @ Wih^T  (K = 640)
      const bf16* br = Wihb + (size_t)(n) * XD;
      const bf16* bz = Wihb + (size_t)(512 + n) * XD;
      const bf16* bn = Wihb + (size_t)(1024 + n) * XD;
      for (int kb = 0; kb < XD; kb += 32) {
        v16bf fr = frag_ld(br, kb, khalf);
        v16bf fz = frag_ld(bz, kb, khalf);
        v16bf fn = frag_ld(bn, kb, khalf);
        v16bf x0 = frag_ld(xa,      kb, khalf);
        v16bf x1 = frag_ld(xa + XD, kb, khalf);
        gr[0] = wmma_bf16(x0, fr, gr[0]); gr[1] = wmma_bf16(x1, fr, gr[1]);
        gz[0] = wmma_bf16(x0, fz, gz[0]); gz[1] = wmma_bf16(x1, fz, gz[1]);
        gn[0] = wmma_bf16(x0, fn, gn[0]); gn[1] = wmma_bf16(x1, fn, gn[1]);
      }
    }
    { // gh = h @ Whh^T for src and tgt  (K = 512); each B-fragment feeds 4 WMMAs
      const bf16* br = Whhb + (size_t)(n) * MEMD;
      const bf16* bz = Whhb + (size_t)(512 + n) * MEMD;
      const bf16* bn = Whhb + (size_t)(1024 + n) * MEMD;
      for (int kb = 0; kb < MEMD; kb += 32) {
        v16bf fr = frag_ld(br, kb, khalf);
        v16bf fz = frag_ld(bz, kb, khalf);
        v16bf fn = frag_ld(bn, kb, khalf);
        v16bf hs0 = frag_ld(sa,        kb, khalf), hs1 = frag_ld(sa + MEMD, kb, khalf);
        v16bf ht0 = frag_ld(ta,        kb, khalf), ht1 = frag_ld(ta + MEMD, kb, khalf);
        sr[0] = wmma_bf16(hs0, fr, sr[0]); sr[1] = wmma_bf16(hs1, fr, sr[1]);
        sz[0] = wmma_bf16(hs0, fz, sz[0]); sz[1] = wmma_bf16(hs1, fz, sz[1]);
        sv[0] = wmma_bf16(hs0, fn, sv[0]); sv[1] = wmma_bf16(hs1, fn, sv[1]);
        tr[0] = wmma_bf16(ht0, fr, tr[0]); tr[1] = wmma_bf16(ht1, fr, tr[1]);
        tz[0] = wmma_bf16(ht0, fz, tz[0]); tz[1] = wmma_bf16(ht1, fz, tz[1]);
        tv[0] = wmma_bf16(ht0, fn, tv[0]); tv[1] = wmma_bf16(ht1, fn, tv[1]);
      }
    }

    float bir = bih[n], biz = bih[512 + n], bin = bih[1024 + n];
    float bhr = bhh[n], bhz = bhh[512 + n], bhn = bhh[1024 + n];
#pragma unroll
    for (int es = 0; es < 2; ++es) {
#pragma unroll
      for (int q = 0; q < 8; ++q) {
        int e = tile * 32 + es * 16 + q + khalf * 8;
        // source update
        int ns_ = src[e];
        float h = mem[(size_t)ns_ * MEMD + n];
        float r = sigmoidf_(gr[es][q] + bir + sr[es][q] + bhr);
        float z = sigmoidf_(gz[es][q] + biz + sz[es][q] + bhz);
        float nv = tanhf(gn[es][q] + bin + r * (sv[es][q] + bhn));
        float upd = (1.f - z) * nv + z * h;
        if (best[ns_] == e) outMem[(size_t)ns_ * MEMD + n] = upd;
        // target update
        int nt_ = tgt[e];
        float h2 = mem[(size_t)nt_ * MEMD + n];
        float r2 = sigmoidf_(gr[es][q] + bir + tr[es][q] + bhr);
        float z2 = sigmoidf_(gz[es][q] + biz + tz[es][q] + bhz);
        float nv2 = tanhf(gn[es][q] + bin + r2 * (tv[es][q] + bhn));
        float upd2 = (1.f - z2) * nv2 + z2 * h2;
        if (best[nt_] == e + B_EV) outMem[(size_t)nt_ * MEMD + n] = upd2;
      }
    }
  }
}

// ---------------------------------------------------------------- launch
extern "C" void kernel_launch(void* const* d_in, const int* in_sizes, int n_in,
                              void* d_out, int out_size, void* d_ws, size_t ws_size,
                              hipStream_t stream) {
  const int*   src   = (const int*)d_in[0];
  const int*   tgt   = (const int*)d_in[1];
  const float* ts    = (const float*)d_in[2];
  const float* edge  = (const float*)d_in[3];
  const float* mem   = (const float*)d_in[4];
  const float* lu    = (const float*)d_in[5];
  const float* freq  = (const float*)d_in[6];
  const float* phase = (const float*)d_in[7];
  const float* msgW  = (const float*)d_in[8];
  const float* msgb  = (const float*)d_in[9];
  const float* Wih   = (const float*)d_in[10];
  const float* Whh   = (const float*)d_in[11];
  const float* bih   = (const float*)d_in[12];
  const float* bhh   = (const float*)d_in[13];

  float* outMem = (float*)d_out;
  float* outLu  = outMem + (size_t)NN * MEMD;

  char* ws = (char*)d_ws;
  size_t o = 0;
  bf16* msgWb = (bf16*)(ws + o); o += (size_t)MSGD * KMSG * 2;
  bf16* Wihb  = (bf16*)(ws + o); o += (size_t)G3 * XD * 2;
  bf16* Whhb  = (bf16*)(ws + o); o += (size_t)G3 * MEMD * 2;
  bf16* xbuf  = (bf16*)(ws + o); o += (size_t)B_EV * XD * 2;
  int*  best  = (int*)(ws + o);  o += (size_t)NN * 4;

  const int totalW = MSGD * KMSG + G3 * XD + G3 * MEMD;
  k_convert_weights<<<(totalW + 255) / 256, 256, 0, stream>>>(msgW, Wih, Whh, msgWb, Wihb, Whhb);
  k_init_best<<<(NN + 255) / 256, 256, 0, stream>>>(best);
  k_copy_mem<<<4096, 256, 0, stream>>>((const float4*)mem, (float4*)outMem, NN * MEMD / 4);
  k_time_enc<<<(B_EV * TIMED + 255) / 256, 256, 0, stream>>>(ts, freq, phase, xbuf);
  k_scatter_pos<<<(2 * B_EV + 255) / 256, 256, 0, stream>>>(src, tgt, best);
  k_msg_gemm<<<B_EV / 64, dim3(32, 8), 64 * KMSG * sizeof(bf16), stream>>>(
      src, tgt, edge, mem, msgWb, msgb, xbuf);
  k_last_update<<<(NN + 255) / 256, 256, 0, stream>>>(ts, lu, best, outLu);
  k_gru_fused<<<B_EV / 32, dim3(32, 8), 32 * (XD + 2 * MEMD) * sizeof(bf16), stream>>>(
      src, tgt, mem, xbuf, Wihb, Whhb, bih, bhh, best, outMem);
}
